// CustomFasterRCNN_83141976916040
// MI455X (gfx1250) — matrix operand order repaired and checked by reading, not compiled, not run on hardware
//
#include <hip/hip_runtime.h>

// ---------------------------------------------------------------------------
// Types for CDNA5 WMMA (wave32): v_wmma_f32_16x16x32_bf16
// ---------------------------------------------------------------------------
typedef __attribute__((ext_vector_type(16))) __bf16 v16bf;
typedef __attribute__((ext_vector_type(8)))  __bf16 v8bf;
typedef __attribute__((ext_vector_type(8)))  float  v8f;
typedef __attribute__((ext_vector_type(4)))  unsigned int v4u;
typedef __attribute__((ext_vector_type(8)))  int v8i;
typedef __attribute__((ext_vector_type(4)))  int v4i;

#define BLK_M 256
#define BLK_N 128
#define TILE_K 32
#define LDSP   40   // padded LDS row pitch in halfs (80B = 64B data + 16B pad)

// ---- staging-path selection (compile-probe driven) ----
#if __has_builtin(__builtin_amdgcn_tensor_load_to_lds) && __has_builtin(__builtin_amdgcn_s_wait_tensorcnt)
#define USE_TDM 1
#else
#define USE_TDM 0
#endif
#if !USE_TDM && __has_builtin(__builtin_amdgcn_global_load_async_to_lds_b128)
#define USE_ASYNC 1
#else
#define USE_ASYNC 0
#endif

static inline int cdiv_h(int a, int b) { return (a + b - 1) / b; }

#if USE_TDM
// Issue one TDM 2D-tile load: rows_tile rows of 32 bf16 (64B) with global row
// stride K elements, into LDS at lds_off with 16B pad after each 64B row.
// OOB rows (>= rows_tensor) are zero-filled by the TDM.
__device__ __forceinline__ void tdm_load_tile(const __bf16* gsrc, unsigned lds_off,
                                              int rows_tile, int rows_tensor, int K)
{
    unsigned long long ga = (unsigned long long)(uintptr_t)gsrc;
    v4u g0;
    g0.x = 1u;                                   // count=1, user descriptor
    g0.y = lds_off;                              // lds_addr (bytes)
    g0.z = (unsigned)(ga & 0xffffffffu);         // global_addr[31:0]
    g0.w = (unsigned)((ga >> 32) & 0x01ffffffu)  // global_addr[56:32]
         | (2u << 30);                           // type = 2 ("image")
    v8i g1;
    g1[0] = (1 << 16)      // data_size = 2B
          | (1 << 20)      // pad_enable
          | (3 << 22)      // pad_interval: 16 DWORDs (64B) between pads
          | (3 << 25);     // pad_amount: 4 DWORDs (16B)
    g1[1] = (32 << 16);                          // tensor_dim0 = 32 elems
    g1[2] = (rows_tensor & 0xffff) << 16;        // tensor_dim1[15:0]
    g1[3] = ((rows_tensor >> 16) & 0xffff)       // tensor_dim1[31:16]
          | (32 << 16);                          // tile_dim0 = 32
    g1[4] = rows_tile & 0xffff;                  // tile_dim1 (tile_dim2 = 0)
    g1[5] = K;                                   // tensor_dim0_stride[31:0]
    g1[6] = 0;
    g1[7] = 0;
    v4i gz = {0, 0, 0, 0};
#if defined(__clang_major__) && __clang_major__ >= 23
    v8i gz8 = {0, 0, 0, 0, 0, 0, 0, 0};
    __builtin_amdgcn_tensor_load_to_lds(g0, g1, gz, gz, gz8, 0);
#else
    __builtin_amdgcn_tensor_load_to_lds(g0, g1, gz, gz, 0);
#endif
}
#endif

#if USE_ASYNC
__device__ __forceinline__ void wait_async0()
{
#if __has_builtin(__builtin_amdgcn_s_wait_asynccnt)
    __builtin_amdgcn_s_wait_asynccnt(0);
#else
    asm volatile("s_wait_asynccnt 0x0" ::: "memory");
#endif
}
#endif

// ---------------------------------------------------------------------------
// Tiled bf16 WMMA GEMM: C[M][N] = act(A[M][K] * Bt[N][K]^T + bias)
// A row-major bf16, Bt row-major bf16 ([N][K]).  N % 128 == 0, K % 32 == 0.
// Block: 256 threads = 8 waves; macro tile 256x128; wave tile 64x64
// (16 v_wmma per K-step per wave, 16 ds_load_b128 -> 1.0 loads/WMMA).
// ---------------------------------------------------------------------------
template <bool OUT_BF16, bool RELU>
__global__ __launch_bounds__(256)
void wmma_gemm(const __bf16* __restrict__ A, const __bf16* __restrict__ Bt,
               const float* __restrict__ bias, void* __restrict__ Cout,
               int M, int N, int K)
{
#if USE_TDM
    __shared__ __bf16 smA[2][BLK_M * LDSP];
    __shared__ __bf16 smB[2][BLK_N * LDSP];
#else
    __shared__ __bf16 smA[1][BLK_M * LDSP];
    __shared__ __bf16 smB[1][BLK_N * LDSP];
#endif

    const int tid    = threadIdx.x;
    const int wave   = tid >> 5;
    const int lane   = tid & 31;
    const int half16 = lane >> 4;
    const int l16    = lane & 15;
    const int wm = wave >> 1;       // 0..3 -> 64-row sub-tile
    const int wn = wave & 1;        // 0..1 -> 64-col sub-tile
    const int Mblk = blockIdx.x * BLK_M;
    const int Nblk = blockIdx.y * BLK_N;

    v8f acc[4][4];
#pragma unroll
    for (int mf = 0; mf < 4; ++mf)
#pragma unroll
        for (int nf = 0; nf < 4; ++nf)
#pragma unroll
            for (int g = 0; g < 8; ++g) acc[mf][nf][g] = 0.f;

    // one K-step of 16 WMMAs from a staged LDS tile pair
    auto compute_step = [&](const __bf16* sa, const __bf16* sb) {
        v16bf bfrag[4];
#pragma unroll
        for (int nf = 0; nf < 4; ++nf) {
            const int col = wn * 64 + nf * 16 + l16;
            union { v16bf v; v8bf h[2]; } u;
            u.h[0] = *(const v8bf*)(sb + col * LDSP + half16 * 8);
            u.h[1] = *(const v8bf*)(sb + col * LDSP + 16 + half16 * 8);
            bfrag[nf] = u.v;
        }
#pragma unroll
        for (int mf = 0; mf < 4; ++mf) {
            const int row = wm * 64 + mf * 16 + l16;
            union { v16bf v; v8bf h[2]; } u;
            u.h[0] = *(const v8bf*)(sa + row * LDSP + half16 * 8);
            u.h[1] = *(const v8bf*)(sa + row * LDSP + 16 + half16 * 8);
#pragma unroll
            for (int nf = 0; nf < 4; ++nf)
                acc[mf][nf] = __builtin_amdgcn_wmma_f32_16x16x32_bf16(
                    false, u.v, false, bfrag[nf],
                    (short)0, acc[mf][nf], false, false);
        }
    };

#if USE_TDM
    // ---- double-buffered TDM staging (TENSORcnt), one issuing wave ----
    const unsigned ldsA0 = (unsigned)(unsigned long long)(uintptr_t)&smA[0][0];
    const unsigned ldsB0 = (unsigned)(unsigned long long)(uintptr_t)&smB[0][0];
    const unsigned abytes = BLK_M * LDSP * 2;
    const unsigned bbytes = BLK_N * LDSP * 2;
    const int nt = K / TILE_K;

    if (wave == 0)
        if (lane == 0) {
            tdm_load_tile(A + (size_t)Mblk * K, ldsA0, BLK_M, M - Mblk, K);
            tdm_load_tile(Bt + (size_t)Nblk * K, ldsB0, BLK_N, N - Nblk, K);
        }
    for (int t = 0; t < nt - 1; ++t) {
        __syncthreads();             // prior compute done -> other buffer free
        if (wave == 0) {
            if (lane == 0) {
                const int k1 = (t + 1) * TILE_K;
                const int b1 = (t + 1) & 1;
                tdm_load_tile(A + (size_t)Mblk * K + k1, ldsA0 + b1 * abytes,
                              BLK_M, M - Mblk, K);
                tdm_load_tile(Bt + (size_t)Nblk * K + k1, ldsB0 + b1 * bbytes,
                              BLK_N, N - Nblk, K);
            }
            __builtin_amdgcn_s_wait_tensorcnt(2);   // tile t finished
        }
        __syncthreads();             // tile t visible to all waves
        compute_step(smA[t & 1], smB[t & 1]);
    }
    __syncthreads();
    if (wave == 0) __builtin_amdgcn_s_wait_tensorcnt(0);
    __syncthreads();
    compute_step(smA[(nt - 1) & 1], smB[(nt - 1) & 1]);

#elif USE_ASYNC
    // ---- async global->LDS staging (ASYNCcnt) ----
    for (int k0 = 0; k0 < K; k0 += TILE_K) {
        __syncthreads();
#pragma unroll
        for (int i = 0; i < 4; ++i) {          // A: 1024 x 16B tasks
            const int task = tid + i * 256;
            const int r = task >> 2, sg = (task & 3) * 8;
            const int gr = min(Mblk + r, M - 1);
            __builtin_amdgcn_global_load_async_to_lds_b128(
                (const __attribute__((address_space(1))) void*)(A + (size_t)gr * K + k0 + sg),
                (__attribute__((address_space(3))) void*)(smA[0] + r * LDSP + sg), 0, 0);
        }
#pragma unroll
        for (int i = 0; i < 2; ++i) {          // B: 512 x 16B tasks
            const int task = tid + i * 256;
            const int r = task >> 2, sg = (task & 3) * 8;
            __builtin_amdgcn_global_load_async_to_lds_b128(
                (const __attribute__((address_space(1))) void*)(Bt + (size_t)(Nblk + r) * K + k0 + sg),
                (__attribute__((address_space(3))) void*)(smB[0] + r * LDSP + sg), 0, 0);
        }
        wait_async0();
        __syncthreads();
        compute_step(smA[0], smB[0]);
    }

#else
    // ---- synchronous staging fallback (branch-free via row clamp) ----
    for (int k0 = 0; k0 < K; k0 += TILE_K) {
        __syncthreads();
#pragma unroll
        for (int task = tid; task < 512; task += 256) {  // A: 512 x 32B tasks
            const int r = task >> 1, sg = (task & 1) * 16;
            const int gr = min(Mblk + r, M - 1);
            const uint4* src = (const uint4*)(A + (size_t)gr * K + k0 + sg);
            uint4* dst = (uint4*)(smA[0] + r * LDSP + sg);
            dst[0] = src[0]; dst[1] = src[1];
            if (k0 + TILE_K < K)
                __builtin_prefetch(A + (size_t)gr * K + k0 + TILE_K + sg, 0, 1);
        }
        {
            const int r = tid >> 1, sg = (tid & 1) * 16;  // B: 256 x 32B tasks
            const uint4* src = (const uint4*)(Bt + (size_t)(Nblk + r) * K + k0 + sg);
            uint4* dst = (uint4*)(smB[0] + r * LDSP + sg);
            dst[0] = src[0]; dst[1] = src[1];
            if (k0 + TILE_K < K)
                __builtin_prefetch(Bt + (size_t)(Nblk + r) * K + k0 + TILE_K + sg, 0, 1);
        }
        __syncthreads();
        compute_step(smA[0], smB[0]);
    }
#endif

    // ---- epilogue: bias + relu, bf16 or f32 out ----
#pragma unroll
    for (int mf = 0; mf < 4; ++mf) {
#pragma unroll
        for (int nf = 0; nf < 4; ++nf) {
            const int col = Nblk + wn * 64 + nf * 16 + l16;
            const float bv = bias ? bias[col] : 0.f;
#pragma unroll
            for (int g = 0; g < 8; ++g) {
                const int row = Mblk + wm * 64 + mf * 16 + g + half16 * 8;
                if (row < M) {
                    float v = acc[mf][nf][g] + bv;
                    if (RELU) v = v > 0.f ? v : 0.f;
                    if (OUT_BF16)
                        ((__bf16*)Cout)[(size_t)row * N + col] = (__bf16)v;
                    else
                        ((float*)Cout)[(size_t)row * N + col] = v;
                }
            }
        }
    }
}

// ---------------------------------------------------------------------------
// conv1: 3->64, stride 2, pad 1, direct (1% of FLOPs). NCHW f32 in, NHWC bf16 out.
// ---------------------------------------------------------------------------
__global__ void conv1_relu(const float* __restrict__ img, const float* __restrict__ W1,
                           __bf16* __restrict__ out)
{
    const int total = 4 * 400 * 400 * 64;
    int t = blockIdx.x * blockDim.x + threadIdx.x;
    if (t >= total) return;
    int co = t & 63;
    int r  = t >> 6;
    int x  = r % 400; r /= 400;
    int y  = r % 400; int b = r / 400;
    float s = 0.f;
#pragma unroll
    for (int ci = 0; ci < 3; ++ci)
#pragma unroll
        for (int kh = 0; kh < 3; ++kh) {
            int iy = y * 2 - 1 + kh;
            if ((unsigned)iy >= 800u) continue;
#pragma unroll
            for (int kw = 0; kw < 3; ++kw) {
                int ix = x * 2 - 1 + kw;
                if ((unsigned)ix >= 800u) continue;
                s += img[((b * 3 + ci) * 800 + iy) * 800 + ix] *
                     W1[((co * 3 + ci) * 3 + kh) * 3 + kw];
            }
        }
    out[t] = (__bf16)(s > 0.f ? s : 0.f);
}

// OIHW f32 -> bf16 [N][K] with K ordered (kh,kw,ci) to match NHWC im2col
__global__ void wperm(const float* __restrict__ W, __bf16* __restrict__ out,
                      int Cout, int Cin)
{
    int t = blockIdx.x * blockDim.x + threadIdx.x;
    int total = Cout * Cin * 9;
    if (t >= total) return;
    int n   = t / (Cin * 9);
    int rem = t % (Cin * 9);
    int ci  = rem / 9;
    int kk  = rem % 9;
    out[(size_t)n * (9 * Cin) + kk * Cin + ci] = (__bf16)W[t];
}

__global__ void f2bf(const float* __restrict__ in, __bf16* __restrict__ out, int n)
{
    int t = blockIdx.x * blockDim.x + threadIdx.x;
    if (t < n) out[t] = (__bf16)in[t];
}

// NHWC bf16 -> im2col rows [Ho*Wo][9*C], K index = (kh*3+kw)*C + ci
__global__ void im2col(const __bf16* __restrict__ in, __bf16* __restrict__ col,
                       int H, int W, int C, int Ho, int Wo, int stride)
{
    int t = blockIdx.x * blockDim.x + threadIdx.x;
    int c8s = C >> 3;
    int total = Ho * Wo * 9 * c8s;
    if (t >= total) return;
    int c8  = t % c8s;
    int r   = t / c8s;
    int kk  = r % 9;
    int pix = r / 9;
    int ox = pix % Wo, oy = pix / Wo;
    int kh = kk / 3,   kw = kk % 3;
    int iy = oy * stride - 1 + kh;
    int ix = ox * stride - 1 + kw;
    uint4 v = make_uint4(0u, 0u, 0u, 0u);
    if ((unsigned)iy < (unsigned)H && (unsigned)ix < (unsigned)W)
        v = *(const uint4*)(in + ((size_t)iy * W + ix) * C + c8 * 8);
    *(uint4*)(col + ((size_t)pix * 9 + kk) * C + c8 * 8) = v;
}

// 1x1 conv heads (obj: Cout=3, reg: Cout=12), f32 out
__global__ void conv1x1_f32(const __bf16* __restrict__ act, const float* __restrict__ Wc,
                            float* __restrict__ out, int Cout)
{
    int t = blockIdx.x * blockDim.x + threadIdx.x;
    int total = 4 * 2500 * Cout;
    if (t >= total) return;
    int c   = t % Cout;
    int pix = t / Cout;
    const __bf16* a = act + (size_t)pix * 256;
    const float*  w = Wc + (size_t)c * 256;
    float s = 0.f;
    for (int k = 0; k < 256; ++k) s += (float)a[k] * w[k];
    out[t] = s;
}

// anchor decode + clip
__global__ void decode_boxes(const float* __restrict__ deltas, float* __restrict__ boxes)
{
    int t = blockIdx.x * blockDim.x + threadIdx.x;
    if (t >= 4 * 7500) return;
    int i  = t % 7500;
    int a  = i % 3;
    int yx = i / 3;
    int px = yx % 50, py = yx / 50;
    const float sizes[3] = {64.f, 128.f, 256.f};
    float h   = sizes[a] * 0.5f;
    float wa  = 2.f * h, ha = 2.f * h;
    float cxa = (px + 0.5f) * 16.f, cya = (py + 0.5f) * 16.f;
    float dx = deltas[t * 4 + 0], dy = deltas[t * 4 + 1];
    float dw = deltas[t * 4 + 2], dh = deltas[t * 4 + 3];
    const float clampv = 4.135166556742356f; // log(1000/16)
    dw = fminf(dw, clampv); dh = fminf(dh, clampv);
    float cx = dx * wa + cxa, cy = dy * ha + cya;
    float w  = __expf(dw) * wa, hh = __expf(dh) * ha;
    float x1 = cx - w * 0.5f, y1 = cy - hh * 0.5f;
    float x2 = cx + w * 0.5f, y2 = cy + hh * 0.5f;
    boxes[t * 4 + 0] = fminf(fmaxf(x1, 0.f), 800.f);
    boxes[t * 4 + 1] = fminf(fmaxf(y1, 0.f), 800.f);
    boxes[t * 4 + 2] = fminf(fmaxf(x2, 0.f), 800.f);
    boxes[t * 4 + 3] = fminf(fmaxf(y2, 0.f), 800.f);
}

// per-image bitonic sort of 8192 (7500 scores padded) in LDS; keep top 2000
__global__ __launch_bounds__(512)
void topk_sort(const float* __restrict__ scores, const float* __restrict__ boxes,
               float* __restrict__ tbox)
{
    __shared__ float s[8192];
    __shared__ int   idx[8192];
    const int b = blockIdx.x;
    for (int i = threadIdx.x; i < 8192; i += 512) {
        s[i]   = (i < 7500) ? scores[b * 7500 + i] : -1e30f;
        idx[i] = i;
    }
    __syncthreads();
    for (int k = 2; k <= 8192; k <<= 1) {
        for (int j = k >> 1; j > 0; j >>= 1) {
            for (int i = threadIdx.x; i < 8192; i += 512) {
                int ixj = i ^ j;
                if (ixj > i) {
                    bool desc = ((i & k) == 0);
                    float si = s[i], sj = s[ixj];
                    if (desc ? (si < sj) : (si > sj)) {
                        s[i] = sj; s[ixj] = si;
                        int tt = idx[i]; idx[i] = idx[ixj]; idx[ixj] = tt;
                    }
                }
            }
            __syncthreads();
        }
    }
    for (int i = threadIdx.x; i < 2000; i += 512) {
        int src = idx[i];
#pragma unroll
        for (int d = 0; d < 4; ++d)
            tbox[((size_t)b * 2000 + i) * 4 + d] = boxes[((size_t)b * 7500 + src) * 4 + d];
    }
}

// greedy NMS on 2000 sorted boxes; emit 1000 proposals per image
__global__ __launch_bounds__(256)
void nms_kernel(const float* __restrict__ tbox, float* __restrict__ props)
{
    __shared__ float bx[2000][4];
    __shared__ float area[2000];
    __shared__ unsigned char supp[2000];
    const int b = blockIdx.x;
    for (int i = threadIdx.x; i < 2000; i += 256) {
#pragma unroll
        for (int d = 0; d < 4; ++d) bx[i][d] = tbox[((size_t)b * 2000 + i) * 4 + d];
        area[i] = (bx[i][2] - bx[i][0]) * (bx[i][3] - bx[i][1]);
        supp[i] = 0;
    }
    __syncthreads();
    for (int i = 0; i < 2000; ++i) {
        if (!supp[i]) {
            float x1 = bx[i][0], y1 = bx[i][1], x2 = bx[i][2], y2 = bx[i][3];
            float ai = area[i];
            for (int j = i + 1 + threadIdx.x; j < 2000; j += 256) {
                float lx = fmaxf(x1, bx[j][0]);
                float ly = fmaxf(y1, bx[j][1]);
                float rx = fminf(x2, bx[j][2]);
                float ry = fminf(y2, bx[j][3]);
                float iw = fmaxf(rx - lx, 0.f), ih = fmaxf(ry - ly, 0.f);
                float inter = iw * ih;
                float iou = inter / (ai + area[j] - inter + 1e-6f);
                if (iou > 0.7f) supp[j] = 1;
            }
        }
        __syncthreads();
    }
    if (threadIdx.x == 0) {
        int cnt = 0;
        for (int i = 0; i < 2000 && cnt < 1000; ++i)
            if (!supp[i]) {
                for (int d = 0; d < 4; ++d)
                    props[((size_t)b * 1000 + cnt) * 4 + d] = bx[i][d];
                ++cnt;
            }
        for (int i = 0; i < 2000 && cnt < 1000; ++i)
            if (supp[i]) {
                for (int d = 0; d < 4; ++d)
                    props[((size_t)b * 1000 + cnt) * 4 + d] = bx[i][d];
                ++cnt;
            }
    }
}

// ROI align: grid (49 cells, 4000 rois), 256 threads = channels.
// Output row layout [c*49 + cell] matches Wfc1's K = c*49 + ph*7 + pw.
__global__ __launch_bounds__(256)
void roi_align(const __bf16* __restrict__ feat, const float* __restrict__ props,
               __bf16* __restrict__ outA)
{
    const int cell = blockIdx.x;   // 0..48
    const int row  = blockIdx.y;   // 0..3999
    const int c    = threadIdx.x;  // 0..255
    const int b    = row / 1000;
    const int ph = cell / 7, pw = cell % 7;
    const float* box = props + (size_t)row * 4;
    const float x1 = box[0] * 0.0625f, y1 = box[1] * 0.0625f;
    const float x2 = box[2] * 0.0625f, y2 = box[3] * 0.0625f;
    const float bw = (x2 - x1) * (1.f / 7.f), bh = (y2 - y1) * (1.f / 7.f);
    const __bf16* f = feat + (size_t)b * 2500 * 256;
    float acc = 0.f;
#pragma unroll
    for (int sy = 0; sy < 2; ++sy) {
        float ys = y1 + ((ph * 2 + sy) + 0.5f) * 0.5f * bh;
        float y0f = fminf(fmaxf(floorf(ys), 0.f), 49.f);
        int y0 = (int)y0f;
        int y1i = min(y0 + 1, 49);
        float wy = fminf(fmaxf(ys - y0f, 0.f), 1.f);
#pragma unroll
        for (int sx = 0; sx < 2; ++sx) {
            float xs = x1 + ((pw * 2 + sx) + 0.5f) * 0.5f * bw;
            float x0f = fminf(fmaxf(floorf(xs), 0.f), 49.f);
            int x0 = (int)x0f;
            int x1i = min(x0 + 1, 49);
            float wx = fminf(fmaxf(xs - x0f, 0.f), 1.f);
            float f00 = (float)f[((size_t)y0 * 50 + x0) * 256 + c];
            float f01 = (float)f[((size_t)y0 * 50 + x1i) * 256 + c];
            float f10 = (float)f[((size_t)y1i * 50 + x0) * 256 + c];
            float f11 = (float)f[((size_t)y1i * 50 + x1i) * 256 + c];
            acc += f00 * (1.f - wy) * (1.f - wx) + f01 * (1.f - wy) * wx +
                   f10 * wy * (1.f - wx) + f11 * wy * wx;
        }
    }
    outA[(size_t)row * 12544 + c * 49 + cell] = (__bf16)(acc * 0.25f);
}

__global__ void write_rows(int* __restrict__ out)
{
    if (threadIdx.x < 4) out[threadIdx.x] = 1000;
}

// ---------------------------------------------------------------------------
// Host launcher
// ---------------------------------------------------------------------------
extern "C" void kernel_launch(void* const* d_in, const int* in_sizes, int n_in,
                              void* d_out, int out_size, void* d_ws, size_t ws_size,
                              hipStream_t stream)
{
    const float* images = (const float*)d_in[0];
    const float* W1   = (const float*)d_in[1];
    const float* W2   = (const float*)d_in[2];
    const float* W3   = (const float*)d_in[3];
    const float* W4   = (const float*)d_in[4];
    const float* Wrpn = (const float*)d_in[5];
    const float* Wobj = (const float*)d_in[6];
    const float* Wreg = (const float*)d_in[7];
    const float* Wfc1 = (const float*)d_in[8];
    const float* b1   = (const float*)d_in[9];
    const float* Wfc2 = (const float*)d_in[10];
    const float* b2   = (const float*)d_in[11];

    char* ws = (char*)d_ws;
    size_t off = 0;
    auto carve = [&](size_t bytes) -> char* {
        char* p = ws + off;
        off += (bytes + 255) & ~(size_t)255;
        return p;
    };

    __bf16* W2b   = (__bf16*)carve((size_t)128 * 576 * 2);
    __bf16* W3b   = (__bf16*)carve((size_t)256 * 1152 * 2);
    __bf16* W4b   = (__bf16*)carve((size_t)256 * 2304 * 2);
    __bf16* Wrpnb = (__bf16*)carve((size_t)256 * 2304 * 2);
    __bf16* Wfc1b = (__bf16*)carve((size_t)1024 * 12544 * 2);
    __bf16* Wfc2b = (__bf16*)carve((size_t)1024 * 1024 * 2);
    __bf16* x1    = (__bf16*)carve((size_t)4 * 400 * 400 * 64 * 2);
    __bf16* x2    = (__bf16*)carve((size_t)4 * 200 * 200 * 128 * 2);
    __bf16* x3    = (__bf16*)carve((size_t)4 * 100 * 100 * 256 * 2);
    __bf16* feat  = (__bf16*)carve((size_t)4 * 2500 * 256 * 2);
    __bf16* tact  = (__bf16*)carve((size_t)4 * 2500 * 256 * 2);
    __bf16* colb  = (__bf16*)carve((size_t)40000 * 576 * 2); // max per-image im2col
    float* scores = (float*)carve((size_t)4 * 7500 * 4);
    float* deltas = (float*)carve((size_t)4 * 7500 * 4 * 4);
    float* boxes  = (float*)carve((size_t)4 * 7500 * 4 * 4);
    float* tbox   = (float*)carve((size_t)4 * 2000 * 4 * 4);
    float* props  = (float*)carve((size_t)4 * 1000 * 4 * 4);
    __bf16* roiA  = (__bf16*)carve((size_t)4000 * 12544 * 2);
    __bf16* h1    = (__bf16*)carve((size_t)4000 * 1024 * 2);

    const int TB = 256;

    // weight conversion
    wperm<<<cdiv_h(128 * 576, TB), TB, 0, stream>>>(W2, W2b, 128, 64);
    wperm<<<cdiv_h(256 * 1152, TB), TB, 0, stream>>>(W3, W3b, 256, 128);
    wperm<<<cdiv_h(256 * 2304, TB), TB, 0, stream>>>(W4, W4b, 256, 256);
    wperm<<<cdiv_h(256 * 2304, TB), TB, 0, stream>>>(Wrpn, Wrpnb, 256, 256);
    f2bf<<<cdiv_h(1024 * 12544, TB), TB, 0, stream>>>(Wfc1, Wfc1b, 1024 * 12544);
    f2bf<<<cdiv_h(1024 * 1024, TB), TB, 0, stream>>>(Wfc2, Wfc2b, 1024 * 1024);

    // conv1 direct
    conv1_relu<<<cdiv_h(4 * 400 * 400 * 64, TB), TB, 0, stream>>>(images, W1, x1);

    // conv2..conv4 + rpn as im2col + WMMA GEMM, per image
    for (int b = 0; b < 4; ++b) {
        // conv2: 64 -> 128, 400 -> 200
        im2col<<<cdiv_h(200 * 200 * 9 * 8, TB), TB, 0, stream>>>(
            x1 + (size_t)b * 400 * 400 * 64, colb, 400, 400, 64, 200, 200, 2);
        wmma_gemm<true, true><<<dim3(cdiv_h(40000, BLK_M), 128 / BLK_N), TB, 0, stream>>>(
            colb, W2b, nullptr, x2 + (size_t)b * 200 * 200 * 128, 40000, 128, 576);
        // conv3: 128 -> 256, 200 -> 100
        im2col<<<cdiv_h(100 * 100 * 9 * 16, TB), TB, 0, stream>>>(
            x2 + (size_t)b * 200 * 200 * 128, colb, 200, 200, 128, 100, 100, 2);
        wmma_gemm<true, true><<<dim3(cdiv_h(10000, BLK_M), 256 / BLK_N), TB, 0, stream>>>(
            colb, W3b, nullptr, x3 + (size_t)b * 100 * 100 * 256, 10000, 256, 1152);
        // conv4: 256 -> 256, 100 -> 50
        im2col<<<cdiv_h(50 * 50 * 9 * 32, TB), TB, 0, stream>>>(
            x3 + (size_t)b * 100 * 100 * 256, colb, 100, 100, 256, 50, 50, 2);
        wmma_gemm<true, true><<<dim3(cdiv_h(2500, BLK_M), 256 / BLK_N), TB, 0, stream>>>(
            colb, W4b, nullptr, feat + (size_t)b * 2500 * 256, 2500, 256, 2304);
        // rpn: 256 -> 256, 50 -> 50 (stride 1)
        im2col<<<cdiv_h(50 * 50 * 9 * 32, TB), TB, 0, stream>>>(
            feat + (size_t)b * 2500 * 256, colb, 50, 50, 256, 50, 50, 1);
        wmma_gemm<true, true><<<dim3(cdiv_h(2500, BLK_M), 256 / BLK_N), TB, 0, stream>>>(
            colb, Wrpnb, nullptr, tact + (size_t)b * 2500 * 256, 2500, 256, 2304);
    }

    // heads
    conv1x1_f32<<<cdiv_h(4 * 2500 * 3, TB), TB, 0, stream>>>(tact, Wobj, scores, 3);
    conv1x1_f32<<<cdiv_h(4 * 2500 * 12, TB), TB, 0, stream>>>(tact, Wreg, deltas, 12);

    // decode + top-2000 + NMS
    decode_boxes<<<cdiv_h(4 * 7500, TB), TB, 0, stream>>>(deltas, boxes);
    topk_sort<<<4, 512, 0, stream>>>(scores, boxes, tbox);
    nms_kernel<<<4, 256, 0, stream>>>(tbox, props);

    // ROI align
    roi_align<<<dim3(49, 4000), 256, 0, stream>>>(feat, props, roiA);

    // FC head (WMMA)
    wmma_gemm<true, true><<<dim3(cdiv_h(4000, BLK_M), 1024 / BLK_N), TB, 0, stream>>>(
        roiA, Wfc1b, b1, h1, 4000, 1024, 12544);
    wmma_gemm<false, true><<<dim3(cdiv_h(4000, BLK_M), 1024 / BLK_N), TB, 0, stream>>>(
        h1, Wfc2b, b2, (float*)d_out, 4000, 1024, 1024);

    // rows output
    write_rows<<<1, 4, 0, stream>>>((int*)d_out + (size_t)4000 * 1024);
}
